// DeformableConv2d_62637803045445
// MI455X (gfx1250) — compile-verified, matrix operand order checked by reading, not run on hardware
//
#include <hip/hip_runtime.h>
#include <hip/hip_bf16.h>

// Problem constants (from reference): B=4, C=128, Co=256, H=W=96, K=3, pad=1
#define HI   96
#define WI   96
#define CI   128
#define CO_  256
#define NB   4
#define KK   9
#define HW   (HI*WI)        // 9216
#define KTOT (KK*CI)        // 1152  (GEMM K, ordered k*C + c)
#define NPIX (NB*HW)        // 36864 (GEMM N)
#define KSTEPS (KTOT/32)    // 36

typedef __attribute__((ext_vector_type(16))) __bf16 v16bf;
typedef __attribute__((ext_vector_type(8)))  float  v8f;

static __device__ __forceinline__ unsigned short f32_to_bf16(float f) {
    unsigned int u = __float_as_uint(f);
    unsigned int r = u + 0x7FFFu + ((u >> 16) & 1u);  // round-to-nearest-even
    return (unsigned short)(r >> 16);
}

// ---------------------------------------------------------------------------
// Kernel 1: offset (18 ch) + mask (9 ch, sigmoid) 3x3 convs, FP32.
// grid = (NPIX/256, 27); blockIdx.y = output channel -> weights are
// block-uniform (scalar loads). x reads are lane-coalesced over w.
// ---------------------------------------------------------------------------
__global__ void conv_offmask_kernel(const float* __restrict__ x,
                                    const float* __restrict__ offw,
                                    const float* __restrict__ offb,
                                    const float* __restrict__ modw,
                                    const float* __restrict__ modb,
                                    float* __restrict__ off_buf,
                                    float* __restrict__ mask_buf) {
    int p  = blockIdx.x * blockDim.x + threadIdx.x;
    int oc = blockIdx.y;
    if (p >= NPIX) return;
    int b = p / HW, hw = p % HW, h = hw / WI, w = hw % WI;

    bool ismask = (oc >= 18);
    const float* wgt = ismask ? (modw + (size_t)(oc - 18) * CI * KK)
                              : (offw + (size_t)oc * CI * KK);
    float acc = ismask ? modb[oc - 18] : offb[oc];

    const float* xb = x + (size_t)b * CI * HW;
    #pragma unroll
    for (int ky = 0; ky < 3; ky++) {
        int iy = h - 1 + ky;
        #pragma unroll
        for (int kx = 0; kx < 3; kx++) {
            int ix  = w - 1 + kx;
            bool v  = (iy >= 0) && (iy < HI) && (ix >= 0) && (ix < WI);
            int idx = v ? (iy * WI + ix) : 0;
            int tap = ky * 3 + kx;
            for (int c = 0; c < CI; c++) {
                float xv = v ? xb[(size_t)c * HW + idx] : 0.f;
                acc = fmaf(xv, wgt[c * KK + tap], acc);
            }
        }
    }
    if (!ismask) off_buf [((size_t)b * 18 + oc      ) * HW + hw] = acc;
    else         mask_buf[((size_t)b * KK + (oc - 18)) * HW + hw] =
                     1.f / (1.f + __expf(-acc));
}

// ---------------------------------------------------------------------------
// Kernel 2: repack weight (Co,C,3,3) fp32 -> Abf[o][k*128+c] bf16
// so the GEMM K ordering matches the im2col column ordering.
// ---------------------------------------------------------------------------
__global__ void repack_w_kernel(const float* __restrict__ weight,
                                unsigned short* __restrict__ Abf) {
    int idx = blockIdx.x * blockDim.x + threadIdx.x;
    if (idx >= CO_ * KTOT) return;
    int o = idx / KTOT, r = idx % KTOT;
    int k = r / CI, c = r % CI;
    Abf[idx] = f32_to_bf16(weight[((size_t)o * CI + c) * KK + k]);
}

// ---------------------------------------------------------------------------
// Kernel 3: bilinear im2col + mask modulation -> bf16 columns.
// col[p][k*128 + c]; corner validity folded into weights (no divergence),
// indices clamped for safety. grid = (NPIX/256, 9 taps).
// ---------------------------------------------------------------------------
__global__ void im2col_kernel(const float* __restrict__ x,
                              const float* __restrict__ off_buf,
                              const float* __restrict__ mask_buf,
                              unsigned short* __restrict__ col) {
    int p = blockIdx.x * blockDim.x + threadIdx.x;
    if (p >= NPIX) return;
    int k  = blockIdx.y;
    int ky = k / 3, kx = k % 3;
    int b = p / HW, hw = p % HW, h = hw / WI, w = hw % WI;

    float dy = off_buf[((size_t)b * 18 + 2 * k    ) * HW + hw];
    float dx = off_buf[((size_t)b * 18 + 2 * k + 1) * HW + hw];
    float m  = mask_buf[((size_t)b * KK + k) * HW + hw];

    float py = (float)(h - 1 + ky) + dy;
    float px = (float)(w - 1 + kx) + dx;
    float fy = floorf(py), fx = floorf(px);
    float ly = py - fy,    lx = px - fx;
    int y0 = (int)fy, x0 = (int)fx, y1 = y0 + 1, x1 = x0 + 1;

    bool vy0 = (y0 >= 0) && (y0 < HI), vy1 = (y1 >= 0) && (y1 < HI);
    bool vx0 = (x0 >= 0) && (x0 < WI), vx1 = (x1 >= 0) && (x1 < WI);
    float w00 = (vy0 && vx0) ? (1.f - ly) * (1.f - lx) * m : 0.f;
    float w01 = (vy0 && vx1) ? (1.f - ly) * lx * m : 0.f;
    float w10 = (vy1 && vx0) ? ly * (1.f - lx) * m : 0.f;
    float w11 = (vy1 && vx1) ? ly * lx * m : 0.f;
    int cy0 = min(max(y0, 0), HI - 1), cy1 = min(max(y1, 0), HI - 1);
    int cx0 = min(max(x0, 0), WI - 1), cx1 = min(max(x1, 0), WI - 1);
    int i00 = cy0 * WI + cx0, i01 = cy0 * WI + cx1;
    int i10 = cy1 * WI + cx0, i11 = cy1 * WI + cx1;

    const float* xb = x + (size_t)b * CI * HW;
    unsigned int* cp = (unsigned int*)(col + ((size_t)p * KTOT + (size_t)k * CI));
    for (int c = 0; c < CI; c += 2) {
        const float* x0p = xb + (size_t)c * HW;
        const float* x1p = x0p + HW;
        float v0 = w00 * x0p[i00] + w01 * x0p[i01] + w10 * x0p[i10] + w11 * x0p[i11];
        float v1 = w00 * x1p[i00] + w01 * x1p[i01] + w10 * x1p[i10] + w11 * x1p[i11];
        cp[c >> 1] = (unsigned int)f32_to_bf16(v0) |
                     ((unsigned int)f32_to_bf16(v1) << 16);
    }
}

// ---------------------------------------------------------------------------
// Kernel 4: WMMA GEMM  out[o][p] = sum_k Abf[o][k]*col[p][k] + bias[o]
// M=256, N=36864, K=1152. WG = 256 thr = 8 waves in a 4(M)x2(N) grid;
// WG tile 256x64, wave tile 64x32 (4x2 fragments of 16x16), K-step 32.
//
// Tile staging uses the CDNA5 async global->LDS path
// (GLOBAL_LOAD_ASYNC_TO_LDS_B128, ASYNCcnt) with double-buffered LDS:
// tile k+1 streams into buf^1 while WMMAs consume buf, one
// s_wait_asynccnt + one workgroup barrier per K-step.
//
// Fragment LDS reads follow the CDNA5 16-bit A/B striping:
//   lanes 0-15  : K {0..7} and {16..23}   (two ds_load_b128)
//   lanes 16-31 : K {8..15} and {24..31}
// ---------------------------------------------------------------------------
__global__ void __launch_bounds__(256) deform_gemm_kernel(
        const unsigned short* __restrict__ Abf,
        const unsigned short* __restrict__ col,
        const float* __restrict__ bias,
        float* __restrict__ out) {
    __shared__ __align__(16) unsigned short Asm[2][256 * 32];  // 2 x 16 KB
    __shared__ __align__(16) unsigned short Bsm[2][64 * 32];   // 2 x  4 KB

    int tid  = threadIdx.x;
    int lane = tid & 31;
    int wave = tid >> 5;
    int wm = wave & 3;             // M offset = 64*wm
    int wn = wave >> 2;            // N offset = 32*wn
    int n0 = blockIdx.x * 64;

    v8f acc[4][2] = {};
    int mrow = lane & 15;
    int kg   = lane >> 4;          // 0: K base {0,16}, 1: K base {8,24}

    int brow = tid >> 2, bch = tid & 3;     // B-tile: 4 threads per pixel-row

    // Issue async global->LDS copies for one 256x32 A tile + 64x32 B tile.
    // LDS address operand = low 32 bits of the flat shared pointer
    // (ISA 10.2).
    auto issue_tile = [&](int buf, int k0) {
        const unsigned short* gA = Abf + (size_t)tid * KTOT + k0;    // row tid
        unsigned lA = (unsigned)(size_t)&Asm[buf][tid * 32];
        #pragma unroll
        for (int i = 0; i < 4; i++) {
            asm volatile("global_load_async_to_lds_b128 %0, %1, off"
                         :: "v"(lA + i * 16), "v"(gA + i * 8) : "memory");
        }
        const unsigned short* gB = col + (size_t)(n0 + brow) * KTOT + k0 + bch * 8;
        unsigned lB = (unsigned)(size_t)&Bsm[buf][brow * 32 + bch * 8];
        asm volatile("global_load_async_to_lds_b128 %0, %1, off"
                     :: "v"(lB), "v"(gB) : "memory");
    };

    issue_tile(0, 0);

    for (int it = 0; it < KSTEPS; ++it) {
        asm volatile("s_wait_asynccnt 0x0" ::: "memory");
        __syncthreads();
        int buf = it & 1;
        if (it + 1 < KSTEPS) issue_tile(buf ^ 1, (it + 1) * 32);

        union Frag { uint4 u[2]; v16bf v; };
        Frag a[4], bf[2];
        #pragma unroll
        for (int f = 0; f < 4; f++) {
            int r = wm * 64 + f * 16 + mrow;
            a[f].u[0] = *(const uint4*)&Asm[buf][r * 32 + kg * 8];
            a[f].u[1] = *(const uint4*)&Asm[buf][r * 32 + 16 + kg * 8];
        }
        #pragma unroll
        for (int g = 0; g < 2; g++) {
            int r = wn * 32 + g * 16 + mrow;
            bf[g].u[0] = *(const uint4*)&Bsm[buf][r * 32 + kg * 8];
            bf[g].u[1] = *(const uint4*)&Bsm[buf][r * 32 + 16 + kg * 8];
        }
        #pragma unroll
        for (int f = 0; f < 4; f++)
            #pragma unroll
            for (int g = 0; g < 2; g++)
                acc[f][g] = __builtin_amdgcn_wmma_f32_16x16x32_bf16(
                    false, a[f].v, false, bf[g].v,
                    (short)0, acc[f][g], false, false);
    }

    // Epilogue: C layout -> VGPR r: lanes 0-15 M=r, lanes 16-31 M=r+8.
    #pragma unroll
    for (int f = 0; f < 4; f++) {
        #pragma unroll
        for (int g = 0; g < 2; g++) {
            int n  = n0 + wn * 32 + g * 16 + (lane & 15);
            int bb = n / HW, hw = n % HW;
            float* op = out + (size_t)bb * CO_ * HW + hw;
            #pragma unroll
            for (int r = 0; r < 8; r++) {
                int o = wm * 64 + f * 16 + kg * 8 + r;
                op[(size_t)o * HW] = acc[f][g][r] + bias[o];
            }
        }
    }
}

// ---------------------------------------------------------------------------
extern "C" void kernel_launch(void* const* d_in, const int* in_sizes, int n_in,
                              void* d_out, int out_size, void* d_ws, size_t ws_size,
                              hipStream_t stream) {
    (void)in_sizes; (void)n_in; (void)out_size; (void)ws_size;
    const float* x      = (const float*)d_in[0];
    const float* weight = (const float*)d_in[1];
    const float* bias   = (const float*)d_in[2];
    const float* offw   = (const float*)d_in[3];
    const float* offb   = (const float*)d_in[4];
    const float* modw   = (const float*)d_in[5];
    const float* modb   = (const float*)d_in[6];
    float* out = (float*)d_out;

    char* ws = (char*)d_ws;
    float* off_buf  = (float*)ws;              ws += (size_t)NB * 18 * HW * sizeof(float);
    float* mask_buf = (float*)ws;              ws += (size_t)NB * KK * HW * sizeof(float);
    unsigned short* Abf = (unsigned short*)ws; ws += (size_t)CO_ * KTOT * sizeof(unsigned short);
    unsigned short* col = (unsigned short*)ws; // NPIX*KTOT bf16 = ~85 MB

    dim3 b256(256);
    conv_offmask_kernel<<<dim3(NPIX / 256, 27), b256, 0, stream>>>(
        x, offw, offb, modw, modb, off_buf, mask_buf);
    repack_w_kernel<<<dim3((CO_ * KTOT) / 256), b256, 0, stream>>>(weight, Abf);
    im2col_kernel<<<dim3(NPIX / 256, KK), b256, 0, stream>>>(
        x, off_buf, mask_buf, col);
    deform_gemm_kernel<<<dim3(NPIX / 64), b256, 0, stream>>>(Abf, col, bias, out);
}
